// EmformerWaveformClassifier_2869038153825
// MI455X (gfx1250) — compile-verified, matrix operand order checked by reading, not run on hardware
//
#include <hip/hip_runtime.h>
#include <math.h>
#include <float.h>

// ---------------------------------------------------------------------------
// Emformer (SEG=20, H=16, L=12, D=F=1024, B=16, T=2000) for MI455X / gfx1250.
// bf16 WMMA (v_wmma_f32_16x16x32_bf16) for dense projections, async
// global->LDS staging (ASYNCcnt) with double-buffered BK=64 stages (16 WMMAs
// per barrier pair), fp32 master activations, fp32 VALU attention
// (block-diagonal 20x20, <1% of FLOPs). Weights pre-transposed to [N,K] bf16
// so both A and B tiles stage as contiguous b128 async transfers.
// ---------------------------------------------------------------------------

typedef __bf16 bf16;
typedef __attribute__((ext_vector_type(16))) __bf16 v16bf;
typedef __attribute__((ext_vector_type(8)))  __bf16 v8bf;
typedef __attribute__((ext_vector_type(8)))  float  v8f;

#define L_LAYERS 12
#define B_BATCH  16
#define T_TIME   2000
#define D_MODEL  1024
#define H_HEADS  16
#define DH       64
#define SEG_LEN  20
#define NSEG     100            // T / SEG
#define M_TOK    32000          // B * T
#define KV_D     2048
#define C_CLS    5

// ---- CDNA5 async global->LDS helpers (inline asm; ISA 08_async_tensor) ----
__device__ __forceinline__ unsigned int lds_off(const void* p)
{
    // addrspacecast generic -> LDS, take the 32-bit LDS byte offset
    return (unsigned int)(size_t)(__attribute__((address_space(3))) const void*)p;
}

__device__ __forceinline__ void async_g2l_b128(unsigned int ldsOff, const bf16* g)
{
    asm volatile("global_load_async_to_lds_b128 %0, %1, off"
                 :: "v"(ldsOff), "v"(g)
                 : "memory");
}

__device__ __forceinline__ void wait_async_le8()
{
    asm volatile("s_wait_asynccnt 0x8" ::: "memory");
}
__device__ __forceinline__ void wait_async_0()
{
    asm volatile("s_wait_asynccnt 0x0" ::: "memory");
}

// ---------------------------------------------------------------------------
// fp32 [K,N] -> bf16 [N,K] tiled transpose+convert (one-time, per layer stack)
// grid: (N/32, K/32, L), block: (32, 8)
// ---------------------------------------------------------------------------
__global__ __launch_bounds__(256) void transpose_cvt_kernel(
    const float* __restrict__ in, bf16* __restrict__ out, int K, int N)
{
    __shared__ float tile[32][33];
    const int n0 = blockIdx.x * 32;
    const int k0 = blockIdx.y * 32;
    const size_t lofs = (size_t)blockIdx.z * (size_t)K * (size_t)N;
    const float* src = in + lofs;
    bf16* dst = out + lofs;
#pragma unroll
    for (int i = 0; i < 4; ++i)
        tile[threadIdx.y + i * 8][threadIdx.x] =
            src[(size_t)(k0 + threadIdx.y + i * 8) * N + n0 + threadIdx.x];
    __syncthreads();
#pragma unroll
    for (int i = 0; i < 4; ++i)
        dst[(size_t)(n0 + threadIdx.y + i * 8) * K + k0 + threadIdx.x] =
            (bf16)tile[threadIdx.x][threadIdx.y + i * 8];
}

// ---------------------------------------------------------------------------
// LayerNorm over D=1024, one block per row, 256 threads x 4 elements
// ---------------------------------------------------------------------------
template<bool OUTBF16>
__global__ __launch_bounds__(256) void ln_kernel(
    const float* __restrict__ x, const float* __restrict__ gamma,
    const float* __restrict__ beta, void* __restrict__ outp)
{
    const int row = blockIdx.x;
    const int tid = threadIdx.x;
    const size_t base = (size_t)row * D_MODEL;

    float v[4];
    float s = 0.f, s2 = 0.f;
#pragma unroll
    for (int i = 0; i < 4; ++i) {
        v[i] = x[base + tid + i * 256];
        s  += v[i];
        s2 += v[i] * v[i];
    }
    __shared__ float r1[256], r2[256];
    r1[tid] = s; r2[tid] = s2;
    __syncthreads();
    for (int st = 128; st > 0; st >>= 1) {
        if (tid < st) { r1[tid] += r1[tid + st]; r2[tid] += r2[tid + st]; }
        __syncthreads();
    }
    const float mu   = r1[0] * (1.f / D_MODEL);
    const float var  = r2[0] * (1.f / D_MODEL) - mu * mu;
    const float rstd = rsqrtf(var + 1e-5f);
#pragma unroll
    for (int i = 0; i < 4; ++i) {
        const int d = tid + i * 256;
        const float y = (v[i] - mu) * rstd * gamma[d] + beta[d];
        if (OUTBF16) ((bf16*)outp)[base + d] = (bf16)y;
        else         ((float*)outp)[base + d] = y;
    }
}

// ---------------------------------------------------------------------------
// bf16 WMMA GEMM: C[M,N] = A[M,K] @ Bt[N,K]^T + bias (+resid) (+relu)
// Block 256 threads = 8 waves, tile 128x128, stage BK=64 (two WMMA K-steps
// per barrier pair), double-buffered LDS, async global->LDS one stage ahead.
// Wave grid 4(M) x 2(N); each wave: 2x4 accumulators -> 16 WMMAs per stage.
// Fragment layout (16-bit operands): lane<16: K 0..7 | 16..23,
// lane>=16: K 8..15 | 24..31 -> two aligned b128 LDS loads per fragment.
// ---------------------------------------------------------------------------
template<bool RELU, bool RESID, bool OUTBF16>
__global__ __launch_bounds__(256) void gemm_bf16_kernel(
    const bf16* __restrict__ A, const bf16* __restrict__ Bt,
    const float* __restrict__ bias, const float* __restrict__ resid,
    void* __restrict__ outp, int Ndim, int Kdim)
{
    constexpr int BM = 128, BN = 128, BK = 64, STR = 72; // 144B rows, 16B-aligned
    constexpr int BUFE = BM * STR;                        // elements per buffer
    __shared__ __align__(16) bf16 sA[2 * BUFE];
    __shared__ __align__(16) bf16 sB[2 * BUFE];

    const int nbn = Ndim / BN;
    const int bm  = blockIdx.x / nbn;
    const int bn  = blockIdx.x % nbn;
    const int tid = threadIdx.x;
    const int lane = tid & 31;
    const int wid  = tid >> 5;
    const int waveM = wid & 3;   // 4 waves along M -> 32 rows each
    const int waveN = wid >> 2;  // 2 waves along N -> 64 cols each
    const int frow  = lane & 15;
    const int fks   = lane >> 4;

    v8f acc[2][4];
#pragma unroll
    for (int i = 0; i < 2; ++i)
#pragma unroll
        for (int j = 0; j < 4; ++j)
            acc[i][j] = (v8f){0.f,0.f,0.f,0.f,0.f,0.f,0.f,0.f};

    const size_t rowA0 = (size_t)bm * BM;
    const int    colB0 = bn * BN;

    // staging: 2 threads per 64-half (128B) row; each thread moves 64B of A
    // and 64B of B per stage as 4+4 async b128 ops
    const int arow  = tid >> 1;          // 0..127 tile row
    const int ahalf = (tid & 1) * 32;    // halves 0 / 32
    const int bcol  = tid >> 1;          // 0..127 tile col (Bt row)
    const int bhalf = (tid & 1) * 32;

    const bf16* agBase = A  + (rowA0 + arow) * (size_t)Kdim + ahalf;
    const bf16* bgBase = Bt + (size_t)(colB0 + bcol) * Kdim + bhalf;

    const unsigned int aOff = lds_off(&sA[arow * STR + ahalf]);
    const unsigned int bOff = lds_off(&sB[bcol * STR + bhalf]);
    constexpr unsigned int BUFB = BUFE * 2;   // buffer stride in bytes

    auto issue_stage = [&](int buf, int kk) {
        const bf16* ag = agBase + kk;
        const bf16* bg = bgBase + kk;
        const unsigned int ab = aOff + (unsigned int)buf * BUFB;
        const unsigned int bb = bOff + (unsigned int)buf * BUFB;
#pragma unroll
        for (int c = 0; c < 4; ++c) async_g2l_b128(ab + 16u * c, ag + 8 * c);
#pragma unroll
        for (int c = 0; c < 4; ++c) async_g2l_b128(bb + 16u * c, bg + 8 * c);
    };

    issue_stage(0, 0);                 // prologue: stage 0 into buffer 0

    int buf = 0;
    for (int k0 = 0; k0 < Kdim; k0 += BK) {
        const bool hasNext = (k0 + BK) < Kdim;
        __syncthreads();               // previous stage fragment reads done block-wide
        if (hasNext) {
            issue_stage(buf ^ 1, k0 + BK);
            wait_async_le8();          // this stage's 8 in-order asyncs landed
        } else {
            wait_async_0();
        }
        __syncthreads();               // all waves' current buffer visible

        const bf16* cA = &sA[buf * BUFE];
        const bf16* cB = &sB[buf * BUFE];

#pragma unroll
        for (int ks = 0; ks < 2; ++ks) {       // two 32-K WMMA steps per stage
            const int kofs = ks * 32;
            v16bf afr[2], bfr[4];
#pragma unroll
            for (int mt = 0; mt < 2; ++mt) {
                const int r = waveM * 32 + mt * 16 + frow;
                v8bf lo = *(const v8bf*)&cA[r * STR + kofs + fks * 8];
                v8bf hi = *(const v8bf*)&cA[r * STR + kofs + 16 + fks * 8];
#pragma unroll
                for (int i = 0; i < 8; ++i) { afr[mt][i] = lo[i]; afr[mt][i + 8] = hi[i]; }
            }
#pragma unroll
            for (int nt = 0; nt < 4; ++nt) {
                const int c = waveN * 64 + nt * 16 + frow;
                v8bf lo = *(const v8bf*)&cB[c * STR + kofs + fks * 8];
                v8bf hi = *(const v8bf*)&cB[c * STR + kofs + 16 + fks * 8];
#pragma unroll
                for (int i = 0; i < 8; ++i) { bfr[nt][i] = lo[i]; bfr[nt][i + 8] = hi[i]; }
            }
#pragma unroll
            for (int mt = 0; mt < 2; ++mt)
#pragma unroll
                for (int nt = 0; nt < 4; ++nt)
                    acc[mt][nt] = __builtin_amdgcn_wmma_f32_16x16x32_bf16(
                        false, afr[mt], false, bfr[nt],
                        (short)0, acc[mt][nt], false, false);
        }

        buf ^= 1;
    }

    // epilogue: bias (+resid) (+relu); C layout: row = e + 8*fks, col = frow
#pragma unroll
    for (int mt = 0; mt < 2; ++mt)
#pragma unroll
        for (int nt = 0; nt < 4; ++nt) {
            const int col = colB0 + waveN * 64 + nt * 16 + frow;
            const float bia = bias[col];
#pragma unroll
            for (int e = 0; e < 8; ++e) {
                const int rowl = waveM * 32 + mt * 16 + fks * 8 + e;
                const size_t off = (rowA0 + rowl) * (size_t)Ndim + col;
                float v = acc[mt][nt][e] + bia;
                if (RELU) v = v > 0.f ? v : 0.f;
                if constexpr (RESID) v += resid[off];
                if constexpr (OUTBF16) ((bf16*)outp)[off] = (bf16)v;
                else                   ((float*)outp)[off] = v;
            }
        }
}

// ---------------------------------------------------------------------------
// Block-diagonal attention: one wave per (b, seg, head). Lanes 0..19 each own
// a query row -> softmax is lane-local.
// q: bf16 [M,1024], kv: bf16 [M,2048] (k | v), ctx out: bf16 [M,1024]
// ---------------------------------------------------------------------------
__global__ __launch_bounds__(256) void attn_kernel(
    const bf16* __restrict__ q, const bf16* __restrict__ kv,
    const int* __restrict__ lengths, bf16* __restrict__ ctx)
{
    const int tid  = threadIdx.x;
    const int wid  = tid >> 5;
    const int lane = tid & 31;
    const int gw   = blockIdx.x * 8 + wid;     // B*NSEG*H = 25600 waves
    const int head = gw % H_HEADS;
    const int rest = gw / H_HEADS;
    const int s    = rest % NSEG;
    const int b    = rest / NSEG;

    const bool act = lane < SEG_LEN;
    const int  ql  = act ? lane : 0;
    const size_t baseTok = (size_t)b * T_TIME + (size_t)s * SEG_LEN;
    const float scaling = 0.125f; // 1/sqrt(64)

    float qr[DH];
    {
        const v8bf* qp = (const v8bf*)(q + (baseTok + ql) * D_MODEL + head * DH);
#pragma unroll
        for (int ch = 0; ch < 8; ++ch) {
            v8bf t = qp[ch];
#pragma unroll
            for (int j = 0; j < 8; ++j) qr[ch * 8 + j] = (float)t[j] * scaling;
        }
    }

    const int len = lengths[b];
    float sc[SEG_LEN];
#pragma unroll 4
    for (int m = 0; m < SEG_LEN; ++m) {
        const v8bf* kp = (const v8bf*)(kv + (baseTok + m) * KV_D + head * DH);
        float dot = 0.f;
#pragma unroll
        for (int ch = 0; ch < 8; ++ch) {
            v8bf t = kp[ch];
#pragma unroll
            for (int j = 0; j < 8; ++j) dot += qr[ch * 8 + j] * (float)t[j];
        }
        const bool valid = (s * SEG_LEN + m) < len;
        sc[m] = valid ? dot : -FLT_MAX;
    }

    float mx = -FLT_MAX;
#pragma unroll
    for (int m = 0; m < SEG_LEN; ++m) mx = fmaxf(mx, sc[m]);
    float sum = 0.f;
#pragma unroll
    for (int m = 0; m < SEG_LEN; ++m) { sc[m] = __expf(sc[m] - mx); sum += sc[m]; }
    const float inv = 1.f / sum;

    float cx[DH];
#pragma unroll
    for (int d = 0; d < DH; ++d) cx[d] = 0.f;
#pragma unroll 4
    for (int m = 0; m < SEG_LEN; ++m) {
        const float p = sc[m] * inv;
        const v8bf* vp = (const v8bf*)(kv + (baseTok + m) * KV_D + D_MODEL + head * DH);
#pragma unroll
        for (int ch = 0; ch < 8; ++ch) {
            v8bf t = vp[ch];
#pragma unroll
            for (int j = 0; j < 8; ++j) cx[ch * 8 + j] += p * (float)t[j];
        }
    }

    if (act) {
        bf16* cp = ctx + (baseTok + lane) * D_MODEL + head * DH;
#pragma unroll
        for (int ch = 0; ch < 8; ++ch) {
            v8bf o;
#pragma unroll
            for (int j = 0; j < 8; ++j) o[j] = (bf16)cx[ch * 8 + j];
            *(v8bf*)(cp + ch * 8) = o;
        }
    }
}

// ---------------------------------------------------------------------------
// pooled = mean over batch of h -> (T,D); out = pooled @ fc_w + fc_b -> (T,5)
// ---------------------------------------------------------------------------
__global__ __launch_bounds__(256) void head_kernel(
    const float* __restrict__ h, const float* __restrict__ fcw,
    const float* __restrict__ fcb, float* __restrict__ out)
{
    const int t   = blockIdx.x;
    const int tid = threadIdx.x;
    __shared__ float pool[D_MODEL];
    __shared__ float rbuf[256];

    for (int d = tid; d < D_MODEL; d += 256) {
        float s = 0.f;
#pragma unroll
        for (int b = 0; b < B_BATCH; ++b)
            s += h[((size_t)b * T_TIME + t) * D_MODEL + d];
        pool[d] = s * (1.f / B_BATCH);
    }
    __syncthreads();

    float acc[C_CLS];
#pragma unroll
    for (int c = 0; c < C_CLS; ++c) acc[c] = 0.f;
    for (int d = tid; d < D_MODEL; d += 256) {
        const float p = pool[d];
#pragma unroll
        for (int c = 0; c < C_CLS; ++c) acc[c] += p * fcw[d * C_CLS + c];
    }
    for (int c = 0; c < C_CLS; ++c) {
        rbuf[tid] = acc[c];
        __syncthreads();
        for (int st = 128; st > 0; st >>= 1) {
            if (tid < st) rbuf[tid] += rbuf[tid + st];
            __syncthreads();
        }
        if (tid == 0) out[(size_t)t * C_CLS + c] = rbuf[0] + fcb[c];
        __syncthreads();
    }
}

// ---------------------------------------------------------------------------
// host: workspace layout + pipeline
// ---------------------------------------------------------------------------
extern "C" void kernel_launch(void* const* d_in, const int* in_sizes, int n_in,
                              void* d_out, int out_size, void* d_ws, size_t ws_size,
                              hipStream_t stream)
{
    (void)in_sizes; (void)n_in; (void)out_size; (void)ws_size;

    const float* x    = (const float*)d_in[0];
    const int*   len  = (const int*)d_in[1];
    const float* lnp  = (const float*)d_in[2];
    const float* Wq   = (const float*)d_in[3];
    const float* bq   = (const float*)d_in[4];
    const float* Wkv  = (const float*)d_in[5];
    const float* bkv  = (const float*)d_in[6];
    const float* Wout = (const float*)d_in[7];
    const float* bout = (const float*)d_in[8];
    const float* W1   = (const float*)d_in[9];
    const float* b1   = (const float*)d_in[10];
    const float* W2   = (const float*)d_in[11];
    const float* b2   = (const float*)d_in[12];
    const float* fcw  = (const float*)d_in[13];
    const float* fcb  = (const float*)d_in[14];
    float* out = (float*)d_out;

    const size_t D = D_MODEL;
    const size_t M = M_TOK;
    const size_t nWq  = (size_t)L_LAYERS * D * D;
    const size_t nWkv = (size_t)L_LAYERS * D * 2 * D;

    char* base = (char*)d_ws;
    size_t off = 0;
    auto take = [&](size_t bytes) -> char* {
        off = (off + 255) & ~(size_t)255;
        char* p = base + off;
        off += bytes;
        return p;
    };
    bf16*  Wq_t   = (bf16*)take(nWq  * 2);   // [l][N=1024][K=1024]
    bf16*  Wkv_t  = (bf16*)take(nWkv * 2);   // [l][N=2048][K=1024]
    bf16*  Wout_t = (bf16*)take(nWq  * 2);
    bf16*  W1_t   = (bf16*)take(nWq  * 2);
    bf16*  W2_t   = (bf16*)take(nWq  * 2);
    float* h      = (float*)take(M * D * 4);
    float* res    = (float*)take(M * D * 4);
    bf16*  hn     = (bf16*)take(M * D * 2);
    bf16*  qb     = (bf16*)take(M * D * 2);
    bf16*  kvb    = (bf16*)take(M * KV_D * 2);
    bf16*  ctxb   = (bf16*)take(M * D * 2);
    bf16*  ff1b   = (bf16*)take(M * D * 2);

    // one-time weight transpose+convert to bf16 [N,K]
    {
        dim3 blk(32, 8);
        transpose_cvt_kernel<<<dim3(1024/32, 1024/32, L_LAYERS), blk, 0, stream>>>(Wq,   Wq_t,   1024, 1024);
        transpose_cvt_kernel<<<dim3(2048/32, 1024/32, L_LAYERS), blk, 0, stream>>>(Wkv,  Wkv_t,  1024, 2048);
        transpose_cvt_kernel<<<dim3(1024/32, 1024/32, L_LAYERS), blk, 0, stream>>>(Wout, Wout_t, 1024, 1024);
        transpose_cvt_kernel<<<dim3(1024/32, 1024/32, L_LAYERS), blk, 0, stream>>>(W1,   W1_t,   1024, 1024);
        transpose_cvt_kernel<<<dim3(1024/32, 1024/32, L_LAYERS), blk, 0, stream>>>(W2,   W2_t,   1024, 1024);
    }

    hipMemcpyAsync(h, x, M * D * 4, hipMemcpyDeviceToDevice, stream);

    const int gN1024 = (M_TOK / 128) * (1024 / 128);  // 2000 blocks
    const int gN2048 = (M_TOK / 128) * (2048 / 128);  // 4000 blocks

    for (int l = 0; l < L_LAYERS; ++l) {
        const bf16*  Wq_l   = Wq_t   + (size_t)l * D * D;
        const bf16*  Wkv_l  = Wkv_t  + (size_t)l * D * 2 * D;
        const bf16*  Wout_l = Wout_t + (size_t)l * D * D;
        const bf16*  W1_l   = W1_t   + (size_t)l * D * D;
        const bf16*  W2_l   = W2_t   + (size_t)l * D * D;
        const float* bq_l   = bq   + (size_t)l * D;
        const float* bkv_l  = bkv  + (size_t)l * 2 * D;
        const float* bout_l = bout + (size_t)l * D;
        const float* b1_l   = b1   + (size_t)l * D;
        const float* b2_l   = b2   + (size_t)l * D;
        const float* ln_l   = lnp  + (size_t)l * 6 * D;

        ln_kernel<true><<<M_TOK, 256, 0, stream>>>(h, ln_l + 0 * D, ln_l + 1 * D, hn);
        gemm_bf16_kernel<false, false, true><<<gN1024, 256, 0, stream>>>(
            hn, Wq_l, bq_l, nullptr, qb, 1024, 1024);
        gemm_bf16_kernel<false, false, true><<<gN2048, 256, 0, stream>>>(
            hn, Wkv_l, bkv_l, nullptr, kvb, 2048, 1024);
        attn_kernel<<<(B_BATCH * NSEG * H_HEADS) / 8, 256, 0, stream>>>(qb, kvb, len, ctxb);
        gemm_bf16_kernel<false, true, false><<<gN1024, 256, 0, stream>>>(
            ctxb, Wout_l, bout_l, h, res, 1024, 1024);
        ln_kernel<true><<<M_TOK, 256, 0, stream>>>(res, ln_l + 2 * D, ln_l + 3 * D, hn);
        gemm_bf16_kernel<true, false, true><<<gN1024, 256, 0, stream>>>(
            hn, W1_l, b1_l, nullptr, ff1b, 1024, 1024);
        gemm_bf16_kernel<false, true, false><<<gN1024, 256, 0, stream>>>(
            ff1b, W2_l, b2_l, res, res, 1024, 1024);
        ln_kernel<false><<<M_TOK, 256, 0, stream>>>(res, ln_l + 4 * D, ln_l + 5 * D, h);
    }

    head_kernel<<<T_TIME, 256, 0, stream>>>(h, fcw, fcb, out);
}
